// _GCN_43258910605772
// MI455X (gfx1250) — matrix operand order repaired and checked by reading, not compiled, not run on hardware
//
#include <hip/hip_runtime.h>
#include <hip/hip_bf16.h>

typedef __bf16 bf16_t;
typedef __attribute__((ext_vector_type(16))) __bf16 v16bf;
typedef __attribute__((ext_vector_type(8)))  __bf16 v8bf;
typedef __attribute__((ext_vector_type(8)))  float  v8f;

#define NN   1024        // nodes
#define TT   12          // time steps
#define BB   32          // batch
#define CC   32          // channels
#define KROWS (BB*TT*CC) // 12288 GEMM columns, stored as rows [(b*12+t)*32+c][n]

// Build a 16-element bf16 fragment from two contiguous 16B loads (off2 = element
// offset of the second half: 16 for k_diffuse A/B fragments, 8 for k_mix W (A) fragments).
static __device__ __forceinline__ v16bf make_frag(const bf16_t* p, int off2) {
    v8bf lo = *(const v8bf*)p;
    v8bf hi = *(const v8bf*)(p + off2);
    v16bf f;
#pragma unroll
    for (int i = 0; i < 8; ++i) { f[i] = lo[i]; f[8 + i] = hi[i]; }
    return f;
}

// ---------------------------------------------------------------------------
// Prep 1: adjT_bf16[a][m][n] = (bf16) adj_a[n][m]   (tiled LDS transpose)
// ---------------------------------------------------------------------------
__global__ __launch_bounds__(256) void k_prep_adj(const float* __restrict__ a0,
                                                  const float* __restrict__ a1,
                                                  const float* __restrict__ a2,
                                                  bf16_t* __restrict__ adjT) {
    __shared__ float tile[32][33];
    const int z = blockIdx.z;
    const float* src = (z == 0) ? a0 : (z == 1) ? a1 : a2;
    bf16_t* dst = adjT + (size_t)z * NN * NN;
    const int mBase = blockIdx.x * 32;
    const int nBase = blockIdx.y * 32;
    const int tx = threadIdx.x, ty = threadIdx.y;
#pragma unroll
    for (int i = 0; i < 4; ++i) {
        int n = nBase + ty + i * 8;
        tile[ty + i * 8][tx] = src[n * NN + mBase + tx];      // coalesced read adj[n][m]
    }
    __syncthreads();
#pragma unroll
    for (int i = 0; i < 4; ++i) {
        int m = mBase + ty + i * 8;
        dst[m * NN + nBase + tx] = (bf16_t)tile[tx][ty + i * 8]; // coalesced write adjT[m][n]
    }
}

// ---------------------------------------------------------------------------
// Prep 2: Xr[((b*12+t)*32+c)][n] = (bf16) x[b][c][n][t]
// ---------------------------------------------------------------------------
__global__ __launch_bounds__(256) void k_prep_x(const float* __restrict__ x,
                                                bf16_t* __restrict__ Xr) {
    __shared__ float s[256 * TT];                    // 12 KB
    const int bc = blockIdx.x;                       // b*32+c
    const int n0 = blockIdx.y * 256;
    const float* src = x + ((size_t)bc * NN + n0) * TT;  // 3072 contiguous floats
    for (int i = threadIdx.x; i < 256 * TT; i += 256) s[i] = src[i];
    __syncthreads();
    const int b = bc >> 5, c = bc & 31;
    for (int i = threadIdx.x; i < 256 * TT; i += 256) {
        int t = i >> 8, n = i & 255;                 // write coalesced along n
        Xr[(size_t)(((b * TT + t) * CC + c)) * NN + n0 + n] = (bf16_t)s[n * TT + t];
    }
}

// ---------------------------------------------------------------------------
// Prep 3: Wbf[g][o][c] = (bf16) W[o][g*32+c]   (7 groups of 32x32)
// ---------------------------------------------------------------------------
__global__ void k_prep_w(const float* __restrict__ W, bf16_t* __restrict__ Wbf) {
    for (int i = threadIdx.x; i < 7 * 32 * 32; i += blockDim.x) {
        int g = i >> 10, rem = i & 1023, o = rem >> 5, c = rem & 31;
        Wbf[i] = (bf16_t)W[o * 224 + g * 32 + c];
    }
}

// ---------------------------------------------------------------------------
// Diffusion GEMM: Yout[col][m] = sum_n adjT[m][n] * Xin[col][n]   (bf16 in/out)
// Block = 8 waves -> 128m x 128col tile; wave = 64m x 32col (4x2 WMMA tiles).
// Software-pipelined: fragments for K-step i+1 are in flight during WMMAs of i.
// ---------------------------------------------------------------------------
__global__ __launch_bounds__(256) void k_diffuse(const bf16_t* __restrict__ adjT,
                                                 const bf16_t* __restrict__ Xin,
                                                 bf16_t* __restrict__ Yout) {
    const int lane = threadIdx.x & 31;
    const int wave = threadIdx.x >> 5;
    const int r = lane & 15, h = lane >> 4;
    const int mBase   = blockIdx.x * 128 + (wave & 1) * 64;
    const int colBase = blockIdx.y * 128 + (wave >> 1) * 32;

    v8f acc[4][2] = {};

    auto loadAset = [&](v16bf* A, int n0) {
#pragma unroll
        for (int mt = 0; mt < 4; ++mt)
            A[mt] = make_frag(adjT + (size_t)(mBase + mt * 16 + r) * NN + n0 + h * 8, 16);
    };
    auto loadBset = [&](v16bf* Bf, int n0) {
#pragma unroll
        for (int ct = 0; ct < 2; ++ct)
            Bf[ct] = make_frag(Xin + (size_t)(colBase + ct * 16 + r) * NN + n0 + h * 8, 16);
    };
    auto mmaset = [&](v16bf* A, v16bf* Bf) {
#pragma unroll
        for (int mt = 0; mt < 4; ++mt)
#pragma unroll
            for (int ct = 0; ct < 2; ++ct)
                acc[mt][ct] = __builtin_amdgcn_wmma_f32_16x16x32_bf16(
                    false, A[mt], false, Bf[ct], (short)0, acc[mt][ct], false, false);
    };

    v16bf A0[4], B0[2], A1[4], B1[2];
    loadAset(A0, 0); loadBset(B0, 0);
    for (int n0 = 0; n0 < NN; n0 += 64) {
        loadAset(A1, n0 + 32); loadBset(B1, n0 + 32);   // n0+32 <= 992, always valid
        mmaset(A0, B0);
        if (n0 + 64 < NN) { loadAset(A0, n0 + 64); loadBset(B0, n0 + 64); }
        mmaset(A1, B1);
    }

    // D: M = v + 8*h, N = r. Consecutive v -> consecutive m -> packed 16B store.
#pragma unroll
    for (int mt = 0; mt < 4; ++mt)
#pragma unroll
        for (int ct = 0; ct < 2; ++ct) {
            bf16_t* pd = Yout + (size_t)(colBase + ct * 16 + r) * NN + mBase + mt * 16 + 8 * h;
            v8bf outv;
#pragma unroll
            for (int v = 0; v < 8; ++v) outv[v] = (bf16_t)acc[mt][ct][v];
            *(v8bf*)pd = outv;
        }
}

// ---------------------------------------------------------------------------
// Channel mix: y[b][o][n][t] = bias[o] + sum_g sum_c Wg[o][c] * Hg[(b,t,c)][n]
// One wave per (b, t, 16-wide n tile); 7 groups x 2 o-tiles = 14 WMMAs in regs.
// Group g+1's strided B-gather is prefetched during group g's WMMAs.
// ---------------------------------------------------------------------------
__global__ __launch_bounds__(256) void k_mix(const bf16_t* __restrict__ Xr,
                                             const bf16_t* __restrict__ H0,
                                             const bf16_t* __restrict__ H1,
                                             const bf16_t* __restrict__ H2,
                                             const bf16_t* __restrict__ H3,
                                             const bf16_t* __restrict__ H4,
                                             const bf16_t* __restrict__ H5,
                                             const bf16_t* __restrict__ Wbf,
                                             const float* __restrict__ bias,
                                             float* __restrict__ y) {
    const int lane = threadIdx.x & 31;
    const int wave = threadIdx.x >> 5;
    const int task = blockIdx.x * 8 + wave;   // 24576 tasks exactly
    const int nt = task & 63;
    const int bt = task >> 6;
    const int b = bt / TT, t = bt % TT;
    const int n0 = nt * 16;
    const int r = lane & 15, h = lane >> 4;

    const bf16_t* grp[7] = {Xr, H0, H1, H2, H3, H4, H5};

    auto loadB = [&](int g) -> v16bf {
        const bf16_t* pb = grp[g] + (size_t)((b * TT + t) * CC + h * 16) * NN + n0 + r;
        v16bf f;
#pragma unroll
        for (int i = 0; i < 16; ++i) f[i] = pb[(size_t)i * NN];
        return f;
    };

    v8f acc0 = {}, acc1 = {};
    v16bf bCur = loadB(0);
#pragma unroll
    for (int g = 0; g < 7; ++g) {
        v16bf bNxt = bCur;
        if (g < 6) bNxt = loadB(g + 1);            // prefetch next group's gather
        const bf16_t* wg = Wbf + g * 1024;
        v16bf a0 = make_frag(wg + r * 32 + h * 16, 8);
        v16bf a1 = make_frag(wg + (16 + r) * 32 + h * 16, 8);
        acc0 = __builtin_amdgcn_wmma_f32_16x16x32_bf16(false, a0, false, bCur, (short)0, acc0, false, false);
        acc1 = __builtin_amdgcn_wmma_f32_16x16x32_bf16(false, a1, false, bCur, (short)0, acc1, false, false);
        bCur = bNxt;
    }
    const int n = n0 + r;
#pragma unroll
    for (int v = 0; v < 8; ++v) {
        int o = 8 * h + v;
        y[(size_t)((b * 32 + o) * NN + n) * TT + t]        = acc0[v] + bias[o];
        y[(size_t)((b * 32 + (o + 16)) * NN + n) * TT + t] = acc1[v] + bias[o + 16];
    }
}

// ---------------------------------------------------------------------------
extern "C" void kernel_launch(void* const* d_in, const int* in_sizes, int n_in,
                              void* d_out, int out_size, void* d_ws, size_t ws_size,
                              hipStream_t stream) {
    const float* x    = (const float*)d_in[0];
    const float* adj0 = (const float*)d_in[1];
    const float* adj1 = (const float*)d_in[2];
    const float* adj2 = (const float*)d_in[3];
    const float* W    = (const float*)d_in[4];
    const float* bias = (const float*)d_in[5];
    float* y = (float*)d_out;

    char* ws = (char*)d_ws;
    const size_t adjT_bytes = (size_t)3 * NN * NN * sizeof(bf16_t);   // 6 MB
    const size_t wbf_bytes  = (size_t)7 * 32 * 32 * sizeof(bf16_t);   // 14 KB
    const size_t buf_bytes  = (size_t)KROWS * NN * sizeof(bf16_t);    // 24 MB

    bf16_t* adjT = (bf16_t*)ws;
    bf16_t* Wbf  = (bf16_t*)(ws + adjT_bytes);
    bf16_t* Xr   = (bf16_t*)(ws + adjT_bytes + wbf_bytes);
    bf16_t* H[6];
    for (int i = 0; i < 6; ++i)
        H[i] = (bf16_t*)(ws + adjT_bytes + wbf_bytes + buf_bytes * (size_t)(i + 1));

    // Prep
    k_prep_adj<<<dim3(NN / 32, NN / 32, 3), dim3(32, 8), 0, stream>>>(adj0, adj1, adj2, adjT);
    k_prep_x  <<<dim3(BB * CC, NN / 256), 256, 0, stream>>>(x, Xr);
    k_prep_w  <<<1, 256, 0, stream>>>(W, Wbf);

    // 6 diffusion GEMMs (hop1, hop2 per adjacency)
    for (int a = 0; a < 3; ++a) {
        const bf16_t* At = adjT + (size_t)a * NN * NN;
        k_diffuse<<<dim3(NN / 128, KROWS / 128), 256, 0, stream>>>(At, Xr,       H[2 * a]);
        k_diffuse<<<dim3(NN / 128, KROWS / 128), 256, 0, stream>>>(At, H[2 * a], H[2 * a + 1]);
    }

    // Fused channel mix + bias + layout transpose to [B, C_OUT, N, T]
    k_mix<<<(BB * TT * (NN / 16)) / 8, 256, 0, stream>>>(
        Xr, H[0], H[1], H[2], H[3], H[4], H[5], Wbf, bias, y);
}
